// MoE_88845693485634
// MI455X (gfx1250) — matrix operand order repaired and checked by reading, not compiled, not run on hardware
//
#include <hip/hip_runtime.h>
#include <hip/hip_bf16.h>
#include <math.h>

#define NUM_EXPERTS 32
#define DIM 1024
#define BATCH 8192

typedef __attribute__((ext_vector_type(16))) _Float16 v16h;
typedef __attribute__((ext_vector_type(8)))  float    v8f;

// ---------------------------------------------------------------------------
// Kernel 1: w_sum[e,i] = sum_o W[e,i,o]
// One wave32 per row (row = e*DIM + i). 8 x float4 coalesced loads per lane.
// Dominant cost of the whole problem: 128 MB streamed once from HBM.
// ---------------------------------------------------------------------------
__global__ __launch_bounds__(256) void wsum_kernel(const float* __restrict__ W,
                                                   float* __restrict__ wsum) {
    const int lane = threadIdx.x & 31;
    const int wave = threadIdx.x >> 5;
    const int row  = blockIdx.x * 8 + wave;              // [0, 32*1024)
    const float4* p = (const float4*)(W + (size_t)row * DIM);

    float s = 0.f;
#pragma unroll
    for (int it = 0; it < 8; ++it) {
        // speculative prefetch two iterations ahead (gfx1250 global_prefetch_b8)
        __builtin_prefetch((const void*)(p + lane + (it + 2) * 32), 0, 0);
        float4 v = p[lane + it * 32];
        s += (v.x + v.y) + (v.z + v.w);
    }
    // wave32 butterfly reduction
#pragma unroll
    for (int off = 16; off > 0; off >>= 1)
        s += __shfl_xor(s, off, 32);
    if (lane == 0) wsum[row] = s;
}

// ---------------------------------------------------------------------------
// Kernel 2: gate logits via v_wmma_f32_16x16x32_f16, then softmax + top-2.
// One wave per 16-token tile. A = x tile (16x32 f16), B = gate_w.T chunk
// (32x16 f16), two N-tiles cover the 32 experts. f32 accumulation.
// ---------------------------------------------------------------------------
__global__ __launch_bounds__(128) void gate_kernel(const float* __restrict__ x,
                                                   const float* __restrict__ gw,
                                                   const float* __restrict__ gb,
                                                   int2*  __restrict__ gidx,
                                                   float2* __restrict__ gval) {
    __shared__ float lds[4][16 * NUM_EXPERTS];           // per-wave logits scratch

    const int lane = threadIdx.x & 31;
    const int wave = threadIdx.x >> 5;
    const int half = lane >> 4;                          // 0: lanes 0-15, 1: 16-31
    const int l    = lane & 15;
    const int m0   = (blockIdx.x * 4 + wave) * 16;       // first token of tile

    const float* arow  = x  + (size_t)(m0 + l) * DIM;    // A: row m0+l of x
    const float* b0row = gw + (size_t)l        * DIM;    // B tile 0: expert l
    const float* b1row = gw + (size_t)(l + 16) * DIM;    // B tile 1: expert l+16

    v8f c0 = {};                                         // experts 0..15
    v8f c1 = {};                                         // experts 16..31

    for (int kk = 0; kk < DIM; kk += 32) {
        // A fragment (16x32, contiguous k order shared with B): this lane holds
        //   K = kk + half*16 .. +15 contiguous in elements 0..15
        v16h a;
        {
            const float* pa0 = arow + kk + half * 16;
#pragma unroll
            for (int j = 0; j < 16; ++j) {
                a[j]     = (_Float16)pa0[j];
            }
        }
        // B fragments (32x16): lane holds column n = lane&15,
        // K = kk + half*16 .. +15 contiguous.
        v16h b0, b1;
        {
            const float* pb0 = b0row + kk + half * 16;
            const float* pb1 = b1row + kk + half * 16;
#pragma unroll
            for (int j = 0; j < 16; ++j) {
                b0[j] = (_Float16)pb0[j];
                b1[j] = (_Float16)pb1[j];
            }
        }
        c0 = __builtin_amdgcn_wmma_f32_16x16x32_f16(false, a, false, b0,
                                                    (short)0, c0, false, false);
        c1 = __builtin_amdgcn_wmma_f32_16x16x32_f16(false, a, false, b1,
                                                    (short)0, c1, false, false);
    }

    // gate bias is constant per column (expert)
    const float bia0 = gb[l];
    const float bia1 = gb[l + 16];
#pragma unroll
    for (int r = 0; r < 8; ++r) { c0[r] += bia0; c1[r] += bia1; }

    // C/D layout: vgpr r, lanes 0-15 -> token r, lanes 16-31 -> token r+8
#pragma unroll
    for (int r = 0; r < 8; ++r) {
        const int m = r + half * 8;
        lds[wave][m * NUM_EXPERTS + l]      = c0[r];
        lds[wave][m * NUM_EXPERTS + 16 + l] = c1[r];
    }
    __syncthreads();

    // 16 lanes per wave each finish one token: softmax + top-2
    if (half == 0) {
        const float* lg = &lds[wave][l * NUM_EXPERTS];
        float mx = lg[0];
#pragma unroll
        for (int e = 1; e < NUM_EXPERTS; ++e) mx = fmaxf(mx, lg[e]);
        float sum = 0.f;
#pragma unroll
        for (int e = 0; e < NUM_EXPERTS; ++e) sum += __expf(lg[e] - mx);

        int i0 = 0; float lv0 = lg[0];
#pragma unroll
        for (int e = 1; e < NUM_EXPERTS; ++e)
            if (lg[e] > lv0) { lv0 = lg[e]; i0 = e; }
        int i1 = (i0 == 0) ? 1 : 0; float lv1 = lg[i1];
#pragma unroll
        for (int e = 0; e < NUM_EXPERTS; ++e)
            if (e != i0 && lg[e] > lv1) { lv1 = lg[e]; i1 = e; }

        const float inv = 1.f / sum;
        const int tok = m0 + l;
        gidx[tok] = make_int2(i0, i1);
        gval[tok] = make_float2(__expf(lv0 - mx) * inv, __expf(lv1 - mx) * inv);
    }
}

// ---------------------------------------------------------------------------
// Kernel 3: out[b,i] = v0*(x*wsum[e0] + bias[e0]) + v1*(x*wsum[e1] + bias[e1])
// One block per token, one float4 per thread. wsum/bias gathers are L2-hot.
// ---------------------------------------------------------------------------
__global__ __launch_bounds__(256) void out_kernel(const float* __restrict__ x,
                                                  const float* __restrict__ wsum,
                                                  const float* __restrict__ ebias,
                                                  const int2*  __restrict__ gidx,
                                                  const float2* __restrict__ gval,
                                                  float* __restrict__ out) {
    const int row = blockIdx.x;
    const int c4  = threadIdx.x;                         // float4 column index

    const int2   idx = gidx[row];
    const float2 val = gval[row];

    const float4 xv = ((const float4*)(x     + (size_t)row   * DIM))[c4];
    const float4 a0 = ((const float4*)(wsum  + (size_t)idx.x * DIM))[c4];
    const float4 a1 = ((const float4*)(wsum  + (size_t)idx.y * DIM))[c4];
    const float4 e0 = ((const float4*)(ebias + (size_t)idx.x * DIM))[c4];
    const float4 e1 = ((const float4*)(ebias + (size_t)idx.y * DIM))[c4];

    float4 o;
    o.x = val.x * fmaf(xv.x, a0.x, e0.x) + val.y * fmaf(xv.x, a1.x, e1.x);
    o.y = val.x * fmaf(xv.y, a0.y, e0.y) + val.y * fmaf(xv.y, a1.y, e1.y);
    o.z = val.x * fmaf(xv.z, a0.z, e0.z) + val.y * fmaf(xv.z, a1.z, e1.z);
    o.w = val.x * fmaf(xv.w, a0.w, e0.w) + val.y * fmaf(xv.w, a1.w, e1.w);

    ((float4*)(out + (size_t)row * DIM))[c4] = o;
}

// ---------------------------------------------------------------------------
extern "C" void kernel_launch(void* const* d_in, const int* in_sizes, int n_in,
                              void* d_out, int out_size, void* d_ws, size_t ws_size,
                              hipStream_t stream) {
    const float* x  = (const float*)d_in[0];   // [8192,1024]
    const float* W  = (const float*)d_in[1];   // [32,1024,1024]
    const float* eb = (const float*)d_in[2];   // [32,1024]
    const float* gw = (const float*)d_in[3];   // [32,1024]
    const float* gb = (const float*)d_in[4];   // [32]
    // d_in[5] = topk (always 2 for this reference)

    float*  wsum = (float*)d_ws;                                          // 128 KB
    int2*   gidx = (int2*)((char*)d_ws + (size_t)NUM_EXPERTS * DIM * 4);  // 64 KB
    float2* gval = (float2*)((char*)gidx + (size_t)BATCH * sizeof(int2)); // 64 KB

    // 1) reduce expert weights over output dim: 32768 rows, 8 waves/block
    wsum_kernel<<<(NUM_EXPERTS * DIM) / 8, 256, 0, stream>>>(W, wsum);

    // 2) WMMA gate GEMM + softmax + top-2: 512 tiles, 4 waves/block
    gate_kernel<<<BATCH / 64, 128, 0, stream>>>(x, gw, gb, gidx, gval);

    // 3) weighted combine
    out_kernel<<<BATCH, 256, 0, stream>>>(x, wsum, eb, gidx, gval, (float*)d_out);
}
